// StructuralEncoder_69741678952581
// MI455X (gfx1250) — matrix operand order repaired and checked by reading, not compiled, not run on hardware
//
#include <hip/hip_runtime.h>
#include <math.h>

// ---------------------------------------------------------------------------
// GATv2 x3 for MI455X (gfx1250, wave32).  Dense projections via
// v_wmma_f32_16x16x32_bf16 with pre-packed B fragments; edge softmax/scatter
// via hardware fp32 atomics (global_atomic_add_f32 / integer atomic max).
// ---------------------------------------------------------------------------

typedef __attribute__((ext_vector_type(16))) __bf16 v16bf;
typedef __attribute__((ext_vector_type(8)))  float  v8f;

#define DM 128        // model dim
#define NH 8          // heads
#define DH 16         // per-head dim
#define NEG_SLOPE 0.2f

// Monotone float <-> ordered-uint mapping for atomic max on floats.
__device__ __forceinline__ unsigned f2ord(float f) {
  unsigned u = __float_as_uint(f);
  return (u & 0x80000000u) ? ~u : (u | 0x80000000u);
}
__device__ __forceinline__ float ord2f(unsigned k) {
  unsigned u = (k & 0x80000000u) ? (k & 0x7fffffffu) : ~k;
  return __uint_as_float(u);
}

__global__ void k_copy(const float* __restrict__ in, float* __restrict__ out,
                       long long n) {
  long long i = blockIdx.x * (long long)blockDim.x + threadIdx.x;
  if (i < n) out[i] = in[i];
}

__global__ void k_zero(float* __restrict__ p, long long n) {
  long long i = blockIdx.x * (long long)blockDim.x + threadIdx.x;
  if (i < n) p[i] = 0.0f;
}

__global__ void k_cvt_bf16(const float* __restrict__ in, __bf16* __restrict__ out,
                           long long n) {
  long long i = blockIdx.x * (long long)blockDim.x + threadIdx.x;
  if (i < n) out[i] = (__bf16)in[i];
}

// ---------------------------------------------------------------------------
// Repack W[128x128] fp32 -> bf16 in WMMA B-fragment order so the GEMM loads
// contiguous 32B per lane:  Wp[((nt*4 + k32)*32 + lane)*16 + j]
// B-frag (32x16 bf16, ISA 7.12.2): lane col n = lane&15, half = lane>>4,
//   VGPR v holds K = (v>=4?16:0) + half*8 + (v&3)*2 (lo 16 bits) / +1 (hi 16 bits).
// ---------------------------------------------------------------------------
__global__ void k_pack_w(const float* __restrict__ W, __bf16* __restrict__ Wp) {
  int t = blockIdx.x * blockDim.x + threadIdx.x;  // 0..16383
  if (t >= DM * DM) return;
  const int j    = t & 15;         // element within lane's 16
  const int lane = (t >> 4) & 31;
  const int k32  = (t >> 9) & 3;   // k-step (K/32)
  const int nt   = t >> 11;        // column tile 0..7
  const int v    = j >> 1;
  const int lo   = j & 1;
  const int half = lane >> 4;
  const int n    = nt * 16 + (lane & 15);
  const int kb   = k32 * 32 + ((v & 4) ? 16 : 0) + half * 8 + (v & 3) * 2 + lo;
  Wp[t] = (__bf16)W[kb * DM + n];
}

// ---------------------------------------------------------------------------
// C[Nn x 128] = Xbf[Nn x 128] @ W + bias[128]   (W pre-packed by k_pack_w)
// Block = 256 threads = 8 waves; block -> one 16-row tile; wave w -> columns
// [16w, 16w+16).  K loop: 4 x v_wmma_f32_16x16x32_bf16.
// A-frag (16x32 bf16): lane m = lane&15, half = lane>>4,
//   VGPR v holds K = (v>=4?16:0) + half*8 + (v&3)*2 (+1 in hi 16 bits)
//   -> two contiguous 16B runs per lane per k-step (coalesces to b128).
// C/D: VGPR v -> row M = v + 8*half, col N = lane&15.
// Store guard hoisted to tile granularity (fast unguarded path for full
// tiles) to avoid per-element exec-mask branching.
// ---------------------------------------------------------------------------
__global__ void __launch_bounds__(256) k_gemm_bias(
    const __bf16* __restrict__ X, const __bf16* __restrict__ Wp,
    const float* __restrict__ bias, float* __restrict__ C, int Nn) {
  const int lane = threadIdx.x & 31;
  const int wave = threadIdx.x >> 5;     // 0..7 -> column tile
  const int half = lane >> 4;
  const int l15  = lane & 15;
  const int tileM = blockIdx.x;

  const int row  = tileM * 16 + l15;
  const int arow = (row < Nn) ? row : 0;   // clamp loads
  const int ncol = wave * 16 + l15;

  const v16bf* Bfrag = (const v16bf*)Wp;   // 32B per lane per k-step

  v8f acc = {};
#pragma unroll
  for (int k32 = 0; k32 < 4; ++k32) {
    const int kk = k32 * 32;
    v16bf a;
#pragma unroll
    for (int v = 0; v < 8; ++v) {
      const int ka = kk + ((v & 4) ? 16 : 0) + half * 8 + (v & 3) * 2;
      a[2 * v]     = X[(long long)arow * DM + ka];
      a[2 * v + 1] = X[(long long)arow * DM + ka + 1];
    }
    const v16bf b = Bfrag[(wave * 4 + k32) * 32 + lane];
    acc = __builtin_amdgcn_wmma_f32_16x16x32_bf16(
        /*neg_a=*/false, a, /*neg_b=*/false, b,
        /*c_mod=*/(short)0, acc, /*reuse_a=*/false, /*reuse_b=*/false);
  }

  const float bv = bias[ncol];
  float* Cbase = C + (long long)(tileM * 16 + 8 * half) * DM + ncol;
  if (tileM * 16 + 16 <= Nn) {
    // full tile: unguarded stores (common path; N % 16 == 0 in practice)
#pragma unroll
    for (int v = 0; v < 8; ++v) Cbase[(long long)v * DM] = acc[v] + bv;
  } else {
#pragma unroll
    for (int v = 0; v < 8; ++v) {
      const int m = tileM * 16 + v + 8 * half;
      if (m < Nn) C[(long long)m * DM + ncol] = acc[v] + bv;
    }
  }
}

// ---------------------------------------------------------------------------
// Per (edge, head): logit = att_h . leaky_relu(xl[src]_h + xr[dst]_h),
// then ordered-int atomicMax into per-(dst,head) running max.
// ---------------------------------------------------------------------------
__global__ void k_edge_logits(const int* __restrict__ ei, long long E,
                              long long Etot, const float* __restrict__ xl,
                              const float* __restrict__ xr,
                              const float* __restrict__ att,
                              float* __restrict__ logits,
                              unsigned* __restrict__ mkey) {
  long long t = blockIdx.x * (long long)blockDim.x + threadIdx.x;
  if (t >= Etot * NH) return;
  const int h = (int)(t & (NH - 1));
  const long long e = t >> 3;
  int s, d;
  if (e < E) { s = ei[e]; d = ei[E + e]; } else { s = d = (int)(e - E); }

  const float4* pl = (const float4*)(xl + (long long)s * DM + h * DH);
  const float4* pr = (const float4*)(xr + (long long)d * DM + h * DH);
  const float4* pa = (const float4*)(att + h * DH);
  float acc = 0.0f;
#pragma unroll
  for (int q = 0; q < 4; ++q) {
    const float4 a = pl[q], b = pr[q], w = pa[q];
    float v0 = a.x + b.x, v1 = a.y + b.y, v2 = a.z + b.z, v3 = a.w + b.w;
    v0 = (v0 > 0.0f) ? v0 : NEG_SLOPE * v0;
    v1 = (v1 > 0.0f) ? v1 : NEG_SLOPE * v1;
    v2 = (v2 > 0.0f) ? v2 : NEG_SLOPE * v2;
    v3 = (v3 > 0.0f) ? v3 : NEG_SLOPE * v3;
    acc += w.x * v0 + w.y * v1 + w.z * v2 + w.w * v3;
  }
  logits[t] = acc;
  atomicMax(&mkey[(long long)d * NH + h], f2ord(acc));
}

// w = exp(logit - max[dst]); denom[dst,h] += w  (hardware fp32 atomic add)
__global__ void k_edge_w(const int* __restrict__ ei, long long E, long long Etot,
                         const unsigned* __restrict__ mkey,
                         float* __restrict__ logits,   // in-place -> w
                         float* __restrict__ denom) {
  long long t = blockIdx.x * (long long)blockDim.x + threadIdx.x;
  if (t >= Etot * NH) return;
  const int h = (int)(t & (NH - 1));
  const long long e = t >> 3;
  int d;
  if (e < E) d = ei[E + e]; else d = (int)(e - E);
  const float m = ord2f(mkey[(long long)d * NH + h]);
  const float w = __expf(logits[t] - m);
  logits[t] = w;
  unsafeAtomicAdd(&denom[(long long)d * NH + h], w);
}

// One 128-thread block per edge: accum[dst, c] += alpha(e, h(c)) * xl[src, c]
__global__ void __launch_bounds__(128) k_scatter(
    const int* __restrict__ ei, long long E, const float* __restrict__ w,
    const float* __restrict__ denom, const float* __restrict__ xl,
    float* __restrict__ accum) {
  const long long e = blockIdx.x;
  const int c = threadIdx.x;     // 0..127
  const int h = c >> 4;
  int s, d;
  if (e < E) { s = ei[e]; d = ei[E + e]; } else { s = d = (int)(e - E); }
  const float alpha = w[e * NH + h] / denom[(long long)d * NH + h];
  unsafeAtomicAdd(&accum[(long long)d * DM + c],
                  alpha * xl[(long long)s * DM + c]);
}

// h = elu(accum + bias); residual += h
__global__ void k_epilogue(const float* __restrict__ accum,
                           const float* __restrict__ bias,
                           float* __restrict__ hbuf, float* __restrict__ res,
                           long long n) {
  long long i = blockIdx.x * (long long)blockDim.x + threadIdx.x;
  if (i >= n) return;
  float v = accum[i] + bias[i & (DM - 1)];
  float hv = (v > 0.0f) ? v : (__expf(v) - 1.0f);
  hbuf[i] = hv;
  res[i] += hv;
}

extern "C" void kernel_launch(void* const* d_in, const int* in_sizes, int n_in,
                              void* d_out, int out_size, void* d_ws,
                              size_t ws_size, hipStream_t stream) {
  const float* x = (const float*)d_in[0];
  // params: d_in[1..18] = {Wl, bl, Wr, br, att, bias} x 3 layers (dict order)
  const int* ei = (const int*)d_in[n_in - 1];  // edge_index [2, E]

  const int Nn = in_sizes[0] / DM;
  const long long E = (long long)in_sizes[n_in - 1] / 2;
  const long long Etot = E + Nn;
  const long long nElems = (long long)Nn * DM;

  // workspace carve-out (256B aligned)
  char* ws = (char*)d_ws;
  size_t off = 0;
  auto alloc = [&](size_t bytes) -> void* {
    void* p = ws + off;
    off = (off + bytes + 255) & ~(size_t)255;
    return p;
  };
  __bf16* xbf    = (__bf16*)alloc((size_t)nElems * 2);
  __bf16* wbfA   = (__bf16*)alloc((size_t)DM * DM * 2);
  __bf16* wbfB   = (__bf16*)alloc((size_t)DM * DM * 2);
  float*  xl     = (float*)alloc((size_t)nElems * 4);
  float*  xr     = (float*)alloc((size_t)nElems * 4);
  float*  logits = (float*)alloc((size_t)Etot * NH * 4);
  unsigned* mkey = (unsigned*)alloc((size_t)Nn * NH * 4);
  float*  denom  = (float*)alloc((size_t)Nn * NH * 4);
  float*  accum  = (float*)alloc((size_t)nElems * 4);
  float*  hbuf   = (float*)alloc((size_t)nElems * 4);
  (void)ws_size; (void)out_size;

  const int T = 256;
  const unsigned gN   = (unsigned)((nElems + T - 1) / T);
  const unsigned gEH  = (unsigned)((Etot * NH + T - 1) / T);
  const unsigned gNH8 = (unsigned)(((long long)Nn * NH + T - 1) / T);
  const unsigned gW   = (unsigned)((DM * DM + T - 1) / T);
  const unsigned gTiles = (unsigned)((Nn + 15) / 16);

  // residual accumulator starts at x
  k_copy<<<gN, T, 0, stream>>>(x, (float*)d_out, nElems);
  k_cvt_bf16<<<gN, T, 0, stream>>>(x, xbf, nElems);

  for (int layer = 0; layer < 3; ++layer) {
    const float* Wl   = (const float*)d_in[1 + layer * 6 + 0];
    const float* bl   = (const float*)d_in[1 + layer * 6 + 1];
    const float* Wr   = (const float*)d_in[1 + layer * 6 + 2];
    const float* br   = (const float*)d_in[1 + layer * 6 + 3];
    const float* att  = (const float*)d_in[1 + layer * 6 + 4];
    const float* bias = (const float*)d_in[1 + layer * 6 + 5];

    k_pack_w<<<gW, T, 0, stream>>>(Wl, wbfA);
    k_pack_w<<<gW, T, 0, stream>>>(Wr, wbfB);

    k_gemm_bias<<<gTiles, 256, 0, stream>>>(xbf, wbfA, bl, xl, Nn);
    k_gemm_bias<<<gTiles, 256, 0, stream>>>(xbf, wbfB, br, xr, Nn);

    k_zero<<<gNH8, T, 0, stream>>>((float*)mkey, (long long)Nn * NH);  // key 0 == -inf
    k_zero<<<gNH8, T, 0, stream>>>(denom, (long long)Nn * NH);
    k_zero<<<gN, T, 0, stream>>>(accum, nElems);

    k_edge_logits<<<gEH, T, 0, stream>>>(ei, E, Etot, xl, xr, att, logits, mkey);
    k_edge_w<<<gEH, T, 0, stream>>>(ei, E, Etot, mkey, logits, denom);
    k_scatter<<<(unsigned)Etot, 128, 0, stream>>>(ei, E, logits, denom, xl, accum);

    k_epilogue<<<gN, T, 0, stream>>>(accum, bias, hbuf, (float*)d_out, nElems);
    k_cvt_bf16<<<gN, T, 0, stream>>>(hbuf, xbf, nElems);  // next layer input
  }
}
